// Encoder_31439160607458
// MI455X (gfx1250) — matrix operand order repaired and checked by reading, not compiled, not run on hardware
//
#include <hip/hip_runtime.h>

typedef __attribute__((ext_vector_type(16))) _Float16 v16h;
typedef __attribute__((ext_vector_type(8)))  float    v8f;

#define B_ 4

// ---------------- copy ----------------
__global__ void k_copy(const float* __restrict__ src, float* __restrict__ dst, int n) {
  int i = blockIdx.x * blockDim.x + threadIdx.x;
  if (i < n) dst[i] = src[i];
}

// ---------------- farthest point sampling ----------------
// One workgroup per batch element; sequential argmax rounds with LDS tree reduce.
__global__ void __launch_bounds__(1024)
k_fps(const float* __restrict__ xyz, int N, int S,
      float* __restrict__ dists, int* __restrict__ fidx, float* __restrict__ new_xyz) {
  int b = blockIdx.x;
  const float* X = xyz + (size_t)b * N * 3;
  float* D = dists + (size_t)b * N;
  __shared__ float rv[1024];
  __shared__ int   ri[1024];
  __shared__ float lp[3];
  int t = threadIdx.x, nt = blockDim.x;
  for (int i = t; i < N; i += nt) D[i] = 1e10f;
  if (t == 0) {
    fidx[b * S + 0] = 0;
    lp[0] = X[0]; lp[1] = X[1]; lp[2] = X[2];
    new_xyz[(size_t)(b * S) * 3 + 0] = X[0];
    new_xyz[(size_t)(b * S) * 3 + 1] = X[1];
    new_xyz[(size_t)(b * S) * 3 + 2] = X[2];
  }
  __syncthreads();
  for (int it = 1; it < S; ++it) {
    float px = lp[0], py = lp[1], pz = lp[2];
    float best = -1.0f; int bi = 0x7fffffff;
    for (int i = t; i < N; i += nt) {
      float dx = X[3 * i + 0] - px, dy = X[3 * i + 1] - py, dz = X[3 * i + 2] - pz;
      float d = dx * dx + dy * dy + dz * dz;
      float dd = D[i]; dd = fminf(dd, d); D[i] = dd;
      if (dd > best) { best = dd; bi = i; }   // keeps lowest index on tie (ascending scan)
    }
    rv[t] = best; ri[t] = bi;
    __syncthreads();
    for (int off = nt >> 1; off > 0; off >>= 1) {
      if (t < off) {
        float ov = rv[t + off]; int oi = ri[t + off];
        if (ov > rv[t] || (ov == rv[t] && oi < ri[t])) { rv[t] = ov; ri[t] = oi; }
      }
      __syncthreads();
    }
    if (t == 0) {
      int j = ri[0];
      fidx[b * S + it] = j;
      float x = X[3 * j], y = X[3 * j + 1], z = X[3 * j + 2];
      lp[0] = x; lp[1] = y; lp[2] = z;
      new_xyz[(size_t)(b * S + it) * 3 + 0] = x;
      new_xyz[(size_t)(b * S + it) * 3 + 1] = y;
      new_xyz[(size_t)(b * S + it) * 3 + 2] = z;
    }
    __syncthreads();
  }
}

// ---------------- ball query: one wave32 per center ----------------
__global__ void k_ball(const float* __restrict__ xyz, const float* __restrict__ new_xyz,
                       int N, int S, int K, float r2, int* __restrict__ bidx) {
  int gid = blockIdx.x * blockDim.x + threadIdx.x;
  int w = gid >> 5, lane = gid & 31;
  if (w >= B_ * S) return;
  int b = w / S, s = w % S;
  const float* X = xyz + (size_t)b * N * 3;
  float cx = new_xyz[(size_t)(b * S + s) * 3 + 0];
  float cy = new_xyz[(size_t)(b * S + s) * 3 + 1];
  float cz = new_xyz[(size_t)(b * S + s) * 3 + 2];
  int* out = bidx + (size_t)(b * S + s) * K;
  int cnt = 0, first = -1;
  for (int base = 0; base < N && cnt < K; base += 32) {
    int i = base + lane;
    bool in = false;
    if (i < N) {
      float dx = X[3 * i] - cx, dy = X[3 * i + 1] - cy, dz = X[3 * i + 2] - cz;
      in = (dx * dx + dy * dy + dz * dz) < r2;
    }
    unsigned mask = (unsigned)__ballot(in);   // wave32: low 32 bits
    if (first < 0 && mask) first = base + __builtin_ctz(mask);
    if (in) {
      int slot = cnt + __popc(mask & ((1u << lane) - 1u));
      if (slot < K) out[slot] = i;
    }
    cnt += __popc(mask);
  }
  if (cnt < K) {
    int fill = (first < 0) ? 0 : first;
    for (int j = cnt + lane; j < K; j += 32) out[j] = fill;
  }
}

// ---------------- group + concat -> f16 padded rows ----------------
__global__ void k_prepare(const float* __restrict__ xyz, const float* __restrict__ new_xyz,
                          const float* __restrict__ fprev, const int* __restrict__ bidx,
                          int N, int S, int K, int Cf, int Cpad, _Float16* __restrict__ Xh) {
  int m = blockIdx.x * blockDim.x + threadIdx.x;
  int M = B_ * S * K;
  if (m >= M) return;
  int k = m % K, s = (m / K) % S, b = m / (K * S);
  int idx = bidx[(size_t)(b * S + s) * K + k];
  const float* p = xyz + (size_t)(b * N + idx) * 3;
  const float* c = new_xyz + (size_t)(b * S + s) * 3;
  _Float16* row = Xh + (size_t)m * Cpad;
  row[0] = (_Float16)(p[0] - c[0]);
  row[1] = (_Float16)(p[1] - c[1]);
  row[2] = (_Float16)(p[2] - c[2]);
  for (int j = 0; j < Cf; ++j)
    row[3 + j] = (_Float16)fprev[((size_t)b * Cf + j) * N + idx];  // fprev [B,Cf,N]
  for (int j = 3 + Cf; j < Cpad; ++j) row[j] = (_Float16)0.f;
}

// ---------------- weight f32 [Cout,Cin] -> f16 [Cout,Cpad] ----------------
__global__ void k_wconv(const float* __restrict__ W, int Cout, int Cin, int Cpad,
                        _Float16* __restrict__ Wh) {
  int i = blockIdx.x * blockDim.x + threadIdx.x;
  if (i >= Cout * Cpad) return;
  int o = i / Cpad, cc = i % Cpad;
  Wh[i] = (cc < Cin) ? (_Float16)W[o * Cin + cc] : (_Float16)0.f;
}

// ---------------- WMMA GEMM: Y[M,Cout] = Xh[M,Cpad] * Wh[Cout,Cpad]^T ----------------
__global__ void k_gemm(const _Float16* __restrict__ Xh, const _Float16* __restrict__ Wh,
                       float* __restrict__ Y, int M, int Cpad, int Cout) {
  int gid = blockIdx.x * blockDim.x + threadIdx.x;
  int w = gid >> 5, lane = gid & 31;
  int tn = Cout >> 4;
  int tiles = (M >> 4) * tn;
  if (w >= tiles) return;                 // wave-uniform: EXEC all-ones at WMMA
  int tm = w / tn, tc = w % tn;
  int l15 = lane & 15, hi = lane >> 4;
  // A 16x32 f16: lane row = tm*16+l15; halves {0..7,16..23} (hi=0) / {8..15,24..31} (hi=1)
  const _Float16* arow = Xh + (size_t)((tm << 4) + l15) * Cpad + hi * 8;
  // B 32x16 f16: lane col = tc*16+l15; halves {0..7,16..23} (hi=0) / {8..15,24..31} (hi=1)
  const _Float16* brow = Wh + (size_t)((tc << 4) + l15) * Cpad + hi * 8;
  v8f acc = {};
  for (int k0 = 0; k0 < Cpad; k0 += 32) {
    union { v16h h; uint4 u[2]; } A, Bf;
    A.u[0]  = *(const uint4*)(arow + k0);
    A.u[1]  = *(const uint4*)(arow + k0 + 16);
    Bf.u[0] = *(const uint4*)(brow + k0);
    Bf.u[1] = *(const uint4*)(brow + k0 + 16);
    acc = __builtin_amdgcn_wmma_f32_16x16x32_f16(false, A.h, false, Bf.h,
                                                 (short)0, acc, false, false);
  }
  int mb = (tm << 4) + (hi << 3);
  int n  = (tc << 4) + l15;
#pragma unroll
  for (int r = 0; r < 8; ++r)
    Y[(size_t)(mb + r) * Cout + n] = acc[r];
}

// ---------------- BN statistics: one workgroup per channel (deterministic) ----------------
__global__ void k_bnstats(const float* __restrict__ Y, int M, int Cout,
                          float* __restrict__ stats) {
  int o = blockIdx.x;
  __shared__ float ss[256], sq[256];
  int t = threadIdx.x;
  float s = 0.f, q = 0.f;
  for (int m = t; m < M; m += blockDim.x) {
    float v = Y[(size_t)m * Cout + o];
    s += v; q += v * v;
  }
  ss[t] = s; sq[t] = q;
  __syncthreads();
  for (int off = blockDim.x >> 1; off > 0; off >>= 1) {
    if (t < off) { ss[t] += ss[t + off]; sq[t] += sq[t + off]; }
    __syncthreads();
  }
  if (t == 0) { stats[2 * o] = ss[0]; stats[2 * o + 1] = sq[0]; }
}

// ---------------- BN + ReLU; emit next layer f16 input or f32 in place ----------------
__global__ void k_bnapply(float* __restrict__ Y, const float* __restrict__ stats,
                          const float* __restrict__ gamma, const float* __restrict__ beta,
                          int M, int Cout, int Cpn, _Float16* __restrict__ Xn, int writeF32) {
  int m = blockIdx.x * blockDim.x + threadIdx.x;
  if (m >= M) return;
  float invM = 1.0f / (float)M;
  float* yr = Y + (size_t)m * Cout;
  if (writeF32) {
    for (int o = 0; o < Cout; ++o) {
      float mean = stats[2 * o] * invM;
      float var  = stats[2 * o + 1] * invM - mean * mean;
      float v = (yr[o] - mean) * rsqrtf(var + 1e-5f) * gamma[o] + beta[o];
      yr[o] = fmaxf(v, 0.f);
    }
  } else {
    _Float16* xr = Xn + (size_t)m * Cpn;
    for (int o = 0; o < Cout; ++o) {
      float mean = stats[2 * o] * invM;
      float var  = stats[2 * o + 1] * invM - mean * mean;
      float v = (yr[o] - mean) * rsqrtf(var + 1e-5f) * gamma[o] + beta[o];
      xr[o] = (_Float16)fmaxf(v, 0.f);
    }
    for (int o = Cout; o < Cpn; ++o) xr[o] = (_Float16)0.f;
  }
}

// ---------------- max over K -> f [B,Cout,S] ----------------
__global__ void k_max(const float* __restrict__ Y, int S, int K, int Cout,
                      float* __restrict__ fout) {
  int gid = blockIdx.x * blockDim.x + threadIdx.x;
  int tot = B_ * S * Cout;
  if (gid >= tot) return;
  int o = gid % Cout;
  int s = (gid / Cout) % S;
  int b = gid / (Cout * S);
  const float* base = Y + ((size_t)(b * S + s) * K) * Cout + o;
  float mx = base[0];
  for (int k = 1; k < K; ++k) mx = fmaxf(mx, base[(size_t)k * Cout]);
  fout[((size_t)b * Cout + o) * S + s] = mx;
}

extern "C" void kernel_launch(void* const* d_in, const int* in_sizes, int n_in,
                              void* d_out, int out_size, void* d_ws, size_t ws_size,
                              hipStream_t stream) {
  (void)in_sizes; (void)n_in; (void)out_size; (void)ws_size;
  const float* pts = (const float*)d_in[0];
  float* out = (float*)d_out;

  // bump allocator over workspace (256B aligned)
  char* wsp = (char*)d_ws;
  auto alloc = [&](size_t bytes) -> void* {
    void* p = (void*)wsp;
    wsp += (bytes + 255) & ~(size_t)255;
    return p;
  };
  float*    dists = (float*)alloc((size_t)B_ * 16384 * 4);
  int*      fidx  = (int*)alloc((size_t)B_ * 1024 * 4);
  int*      bidx  = (int*)alloc((size_t)B_ * 1024 * 48 * 4);
  _Float16* XhA   = (_Float16*)alloc((size_t)196608 * 32 * 2);   // 12.6 MB
  _Float16* XhB   = (_Float16*)alloc((size_t)196608 * 32 * 2);   // 12.6 MB
  float*    Y     = (float*)alloc((size_t)196608 * 64 * 4);      // 50.3 MB
  _Float16* Wh    = (_Float16*)alloc((size_t)256 * 256 * 2);
  float*    stats = (float*)alloc((size_t)2 * 256 * 4);

  auto pad32 = [](int c) { return (c + 31) & ~31; };

  // output 0: pts passthrough
  k_copy<<<(196608 + 255) / 256, 256, 0, stream>>>(pts, out, 196608);
  size_t off = 196608;

  const int   Ss[4]    = {1024, 512, 256, 256};
  const int   Ks[4]    = {48, 32, 32, 8};
  const float Rs[4]    = {1.6f, 3.2f, 6.4f, 10.0f};
  const int   NL[4]    = {3, 2, 2, 3};
  const int   CH[4][4] = {{3, 16, 32, 64}, {67, 64, 64, 0}, {67, 128, 128, 0}, {131, 128, 256, 256}};

  const float* xyz_cur = pts;
  int N = 16384;
  const float* f_cur = nullptr;
  int Cf = 0;
  int pi = 1;  // d_in index: params flattened (W, gamma, beta) per layer per stage

  for (int st = 0; st < 4; ++st) {
    int S = Ss[st], K = Ks[st];
    float r = Rs[st];
    float* xyz_new = out + off; off += (size_t)B_ * S * 3;

    k_fps<<<B_, 1024, 0, stream>>>(xyz_cur, N, S, dists, fidx, xyz_new);

    int nthr = B_ * S * 32;  // one wave per center
    k_ball<<<(nthr + 255) / 256, 256, 0, stream>>>(xyz_cur, xyz_new, N, S, K, r * r, bidx);

    int M = B_ * S * K;
    int Cin = CH[st][0];
    int Cpad = pad32(Cin);
    k_prepare<<<(M + 255) / 256, 256, 0, stream>>>(xyz_cur, xyz_new, f_cur, bidx,
                                                   N, S, K, Cf, Cpad, XhA);
    _Float16* Xc = XhA;
    _Float16* Xn = XhB;
    int Cout = 0;
    for (int l = 0; l < NL[st]; ++l) {
      Cout = CH[st][l + 1];
      const float* Wt    = (const float*)d_in[pi++];
      const float* gamma = (const float*)d_in[pi++];
      const float* beta  = (const float*)d_in[pi++];
      k_wconv<<<(Cout * Cpad + 255) / 256, 256, 0, stream>>>(Wt, Cout, Cin, Cpad, Wh);
      int tiles = (M >> 4) * (Cout >> 4);
      k_gemm<<<(tiles * 32 + 255) / 256, 256, 0, stream>>>(Xc, Wh, Y, M, Cpad, Cout);
      k_bnstats<<<Cout, 256, 0, stream>>>(Y, M, Cout, stats);
      bool last = (l == NL[st] - 1);
      int Cpn = last ? 0 : pad32(Cout);
      k_bnapply<<<(M + 255) / 256, 256, 0, stream>>>(Y, stats, gamma, beta, M, Cout,
                                                     Cpn, Xn, last ? 1 : 0);
      if (!last) {
        _Float16* tmp = Xc; Xc = Xn; Xn = tmp;
        Cin = Cout; Cpad = Cpn;
      }
    }
    float* f_new = out + off; off += (size_t)B_ * Cout * S;
    k_max<<<(B_ * S * Cout + 255) / 256, 256, 0, stream>>>(Y, S, K, Cout, f_new);

    xyz_cur = xyz_new;
    N = S;
    f_cur = f_new;
    Cf = Cout;
  }
}
